// LSTMEncoderDecoder_26525718020602
// MI455X (gfx1250) — matrix operand order repaired and checked by reading, not compiled, not run on hardware
//
#include <hip/hip_runtime.h>
#include <hip/hip_bf16.h>

typedef __attribute__((ext_vector_type(8)))  float   v8f;
typedef __attribute__((ext_vector_type(16))) __bf16  v16bf;
typedef __attribute__((ext_vector_type(8)))  __bf16  bf16x8;

#define HROW 264  // padded LDS row stride (elements) -> bank-conflict-free loads

union FragU { v16bf v; bf16x8 h[2]; };

__device__ __forceinline__ float sigm(float x) { return 1.0f / (1.0f + __expf(-x)); }
__device__ __forceinline__ float tanh_fast(float x) {
  float e = __expf(2.0f * x);
  return (e - 1.0f) / (e + 1.0f);
}

// B fragment (16x32 K-major tile), packed layout: [half][lane][8 bf16] = 1KB block
__device__ __forceinline__ v16bf load_fragB(const __bf16* blk, int lane) {
  FragU f;
  f.h[0] = *(const bf16x8*)(blk + lane * 8);
  f.h[1] = *(const bf16x8*)(blk + 256 + lane * 8);
  return f.v;
}

// A fragment from LDS row-major h: lane holds row lane%16; K = e + 8*(lane/16) + 8*(e>=8)
__device__ __forceinline__ v16bf load_fragA_lds(const __bf16* p) {
  FragU f;
  f.h[0] = *(const bf16x8*)(p);
  f.h[1] = *(const bf16x8*)(p + 16);
  return f.v;
}

// A fragment for the padded input GEMM: stored pre-swizzled, 16 contiguous bf16/lane
__device__ __forceinline__ v16bf load_fragA_x(const __bf16* p) {
  FragU f;
  f.h[0] = *(const bf16x8*)(p);
  f.h[1] = *(const bf16x8*)(p + 8);
  return f.v;
}

// -------- prep: pack W[N][Kw] (f32) into bf16 WMMA B-fragment blocks --------
// out block b = kt*NT + nt; within block: [half(256)][lane(8)][ei]
// element (lane,e): n = nt*16 + lane%16 ; k = kt*32 + 8*(lane/16) + e + 8*(e>=8) ; 0 if k>=Kw
__global__ void pack_b_frag(const float* __restrict__ W, __bf16* __restrict__ out,
                            int NT, int KTB, int Kw) {
  int idx = blockIdx.x * blockDim.x + threadIdx.x;
  int total = KTB * NT * 512;
  if (idx >= total) return;
  int b    = idx >> 9;
  int w    = idx & 511;
  int half = w >> 8;
  int lane = (w >> 3) & 31;
  int ei   = w & 7;
  int e    = half * 8 + ei;
  int nt   = b % NT;
  int kt   = b / NT;
  int n    = nt * 16 + (lane & 15);
  int k    = kt * 32 + 8 * (lane >> 4) + e + 8 * half;
  float v  = (k < Kw) ? W[n * Kw + k] : 0.0f;
  out[idx] = (__bf16)v;
}

// pack Wih (K padded to 32) AND fold combined bias (bih+bhh) into the k==7 row.
// The x A-fragment carries a constant 1.0 at k==7, so gates += 1.0*bias[n].
__global__ void pack_wih_frag(const float* __restrict__ W,
                              const float* __restrict__ bih,
                              const float* __restrict__ bhh,
                              __bf16* __restrict__ out, int Kw) {
  int idx = blockIdx.x * blockDim.x + threadIdx.x;   // 64 blocks * 512
  if (idx >= 64 * 512) return;
  int b    = idx >> 9;     // nt (KTB == 1)
  int w    = idx & 511;
  int half = w >> 8;
  int lane = (w >> 3) & 31;
  int ei   = w & 7;
  int e    = half * 8 + ei;
  int n    = b * 16 + (lane & 15);
  int k    = 16 * (lane >> 4) + e;
  float v  = 0.0f;
  if (k < Kw)      v = W[n * Kw + k];
  else if (k == 7) v = bih[n] + bhh[n];
  out[idx] = (__bf16)v;
}

// -------- one LSTM cell step: gates = h@Whh^T + [x,1]@[Wih,b]^T ; update c,h --------
// wave role: mb = row block (16 rows), jg = 32-column group (0..7); 2 j-tiles x 4 gates
__device__ __forceinline__ void lstm_cell_step(
    const __bf16* __restrict__ whh_frag,  // [8*64][512]
    const __bf16* __restrict__ wih_frag,  // [64][512] (zero-padded K, bias at k=7)
    const __bf16* h_in,                   // LDS [32][HROW]
    __bf16*       h_out,                  // LDS [32][HROW]
    const __bf16* xfrag,                  // LDS [2][32][16], 1.0 at k=7
    v8f*          creg,                   // [2] per-wave cell state
    int mb, int jg, int lane, int lcol, int lhalf)
{
  v8f acc[4][2];
  const v8f zero = {0.f, 0.f, 0.f, 0.f, 0.f, 0.f, 0.f, 0.f};

  // input + bias contribution (D_in padded to K=32; bias rides on k==7).
  // SRC2 = constant zero vector -> inline 0, no accumulator init movs.
  {
    v16bf ax = load_fragA_x(xfrag + mb * 512 + lane * 16);
#pragma unroll
    for (int g = 0; g < 4; ++g)
#pragma unroll
      for (int jt = 0; jt < 2; ++jt) {
        int nt = g * 16 + jg * 2 + jt;
        v16bf bx = load_fragB(wih_frag + nt * 512, lane);
        acc[g][jt] = __builtin_amdgcn_wmma_f32_16x16x32_bf16(
            false, ax, false, bx, (short)0, zero, false, false);
      }
  }

  // recurrent contribution: K = 256 in 8 steps of 32.
  // unroll 1: bound live ranges (A + 8 B frags) -> no spill; 16 waves hide latency.
#pragma unroll 1
  for (int kt = 0; kt < 8; ++kt) {
    v16bf a = load_fragA_lds(h_in + (mb * 16 + lcol) * HROW + kt * 32 + 8 * lhalf);
#pragma unroll
    for (int g = 0; g < 4; ++g)
#pragma unroll
      for (int jt = 0; jt < 2; ++jt) {
        int nt = g * 16 + jg * 2 + jt;
        v16bf b = load_fragB(whh_frag + (kt * 64 + nt) * 512, lane);
        acc[g][jt] = __builtin_amdgcn_wmma_f32_16x16x32_bf16(
            false, a, false, b, (short)0, acc[g][jt], false, false);
      }
  }

  // activations + state update (gate order i,f,g,o); c stays in registers
  int rowb = mb * 16 + 8 * lhalf;
#pragma unroll
  for (int jt = 0; jt < 2; ++jt) {
    int col = jg * 32 + jt * 16 + lcol;
#pragma unroll
    for (int r = 0; r < 8; ++r) {
      float gi = acc[0][jt][r];
      float gf = acc[1][jt][r];
      float gg = acc[2][jt][r];
      float go = acc[3][jt][r];
      float cn = sigm(gf) * creg[jt][r] + sigm(gi) * tanh_fast(gg);
      float hn = sigm(go) * tanh_fast(cn);
      creg[jt][r] = cn;
      h_out[(rowb + r) * HROW + col] = (__bf16)hn;
    }
  }
}

__global__ __launch_bounds__(512, 1) void lstm_encdec_kernel(
    const float* __restrict__ source,
    const __bf16* __restrict__ whh_enc_frag,
    const __bf16* __restrict__ wih_enc_frag,
    const __bf16* __restrict__ whh_dec_frag,
    const __bf16* __restrict__ wih_dec_frag,
    const __bf16* __restrict__ w1_frag,
    const float* __restrict__ b1, const float* __restrict__ W2,
    const float* __restrict__ b2,
    float* __restrict__ out)
{
  __shared__ __bf16 h_buf[2][32 * HROW];  // double-buffered hidden state (bf16)
  __shared__ __bf16 r_buf[32 * HROW];     // relu(h@W1^T+b1)
  __shared__ __bf16 xfrag[1024];          // pre-swizzled A-fragment of x (2 mblocks)
  __shared__ float  b1s[256];
  __shared__ float  pred_part[512];

  const int tid   = threadIdx.x;
  const int lane  = tid & 31;
  const int wid   = tid >> 5;     // 0..15
  const int mb    = wid >> 3;     // row block (16 rows each)
  const int jg    = wid & 7;      // 32-column group within each gate
  const int lcol  = lane & 15;
  const int lhalf = lane >> 4;
  const int b0    = blockIdx.x * 32;

  // init xfrag: zeros everywhere except constant 1.0 at (lanes 0-15, e==7) -> k==7
  for (int i = tid; i < 1024; i += 512) {
    int xl = (i >> 4) & 31, xe = i & 15;
    xfrag[i] = (xl < 16 && xe == 7) ? (__bf16)1.0f : (__bf16)0.0f;
  }
  if (tid < 256) b1s[tid] = b1[tid];
  for (int i = tid; i < 32 * HROW; i += 512) h_buf[0][i] = (__bf16)0.0f;

  v8f creg[2];
#pragma unroll
  for (int jt = 0; jt < 2; ++jt)
#pragma unroll
    for (int r = 0; r < 8; ++r) creg[jt][r] = 0.0f;

  int cur = 0;
  __syncthreads();

  // ---------------- encoder: 64 recurrent steps ----------------
  for (int t = 0; t < 64; ++t) {
    if (tid < 32) {  // stage x_t into padded A-fragment (rows = batch tile)
      const float4 xv = *(const float4*)(source + ((size_t)(b0 + tid) * 64 + t) * 4);
      __bf16* dst = &xfrag[(tid >> 4) * 512 + (tid & 15) * 16];
      dst[0] = (__bf16)xv.x; dst[1] = (__bf16)xv.y;
      dst[2] = (__bf16)xv.z; dst[3] = (__bf16)xv.w;
    }
    __syncthreads();
    lstm_cell_step(whh_enc_frag, wih_enc_frag,
                   h_buf[cur], h_buf[cur ^ 1], xfrag, creg,
                   mb, jg, lane, lcol, lhalf);
    cur ^= 1;
    __syncthreads();
  }

  // decoder x0 = 0 (clear stale encoder inputs; keeps the 1.0 at e==7)
  if (tid < 32) {
    __bf16* dst = &xfrag[(tid >> 4) * 512 + (tid & 15) * 16];
    dst[0] = dst[1] = dst[2] = dst[3] = (__bf16)0.0f;
  }
  __syncthreads();

  // ---------------- decoder: 12 autoregressive steps ----------------
  for (int s = 0; s < 12; ++s) {
    lstm_cell_step(whh_dec_frag, wih_dec_frag,
                   h_buf[cur], h_buf[cur ^ 1], xfrag, creg,
                   mb, jg, lane, lcol, lhalf);
    cur ^= 1;
    __syncthreads();

    // head: r = relu(h_new @ W1^T + b1)   (M=32, K=256, N=256): 2 tiles/wave
    {
      v8f hacc[2];
#pragma unroll
      for (int jt = 0; jt < 2; ++jt) {
        float b = b1s[jg * 32 + jt * 16 + lcol];
        v8f a = {b, b, b, b, b, b, b, b};
        hacc[jt] = a;
      }
#pragma unroll 1
      for (int kt = 0; kt < 8; ++kt) {
        v16bf a = load_fragA_lds(h_buf[cur] + (mb * 16 + lcol) * HROW + kt * 32 + 8 * lhalf);
#pragma unroll
        for (int jt = 0; jt < 2; ++jt) {
          int nt = jg * 2 + jt;
          v16bf b = load_fragB(w1_frag + (kt * 16 + nt) * 512, lane);
          hacc[jt] = __builtin_amdgcn_wmma_f32_16x16x32_bf16(
              false, a, false, b, (short)0, hacc[jt], false, false);
        }
      }
      int rowb = mb * 16 + 8 * lhalf;
#pragma unroll
      for (int jt = 0; jt < 2; ++jt) {
        int col = jg * 32 + jt * 16 + lcol;
#pragma unroll
        for (int r = 0; r < 8; ++r) {
          float v = hacc[jt][r];
          v = v > 0.0f ? v : 0.0f;
          r_buf[(rowb + r) * HROW + col] = (__bf16)v;
        }
      }
    }
    __syncthreads();

    // pred = r @ W2^T + b2  (N=2): per-row partial dots, 8 partials/row (tid<256)
    if (tid < 256) {
      int m = tid >> 3, sg = tid & 7;
      const __bf16* rr  = &r_buf[m * HROW + sg * 32];
      const float*  w2a = W2 + sg * 32;
      const float*  w2b = W2 + 256 + sg * 32;
      float a0 = 0.0f, a1 = 0.0f;
#pragma unroll 8
      for (int k = 0; k < 32; ++k) {
        float rv = (float)rr[k];
        a0 += rv * w2a[k];
        a1 += rv * w2b[k];
      }
      pred_part[(m * 8 + sg) * 2 + 0] = a0;
      pred_part[(m * 8 + sg) * 2 + 1] = a1;
    }
    __syncthreads();
    if (tid < 64) {  // finalize, write output, feed prediction back as next x
      int m = tid >> 1, o = tid & 1;
      float sum = b2[o];
#pragma unroll
      for (int p = 0; p < 8; ++p) sum += pred_part[(m * 8 + p) * 2 + o];
      out[((size_t)(b0 + m)) * 24 + s * 2 + o] = sum;
      xfrag[(m >> 4) * 512 + (m & 15) * 16 + o] = (__bf16)sum;
    }
    __syncthreads();
  }
}

extern "C" void kernel_launch(void* const* d_in, const int* in_sizes, int n_in,
                              void* d_out, int out_size, void* d_ws, size_t ws_size,
                              hipStream_t stream) {
  const float* source  = (const float*)d_in[0];
  const float* enc_Wih = (const float*)d_in[1];
  const float* enc_Whh = (const float*)d_in[2];
  const float* enc_bih = (const float*)d_in[3];
  const float* enc_bhh = (const float*)d_in[4];
  const float* dec_Wih = (const float*)d_in[5];
  const float* dec_Whh = (const float*)d_in[6];
  const float* dec_bih = (const float*)d_in[7];
  const float* dec_bhh = (const float*)d_in[8];
  const float* W1      = (const float*)d_in[9];
  const float* b1      = (const float*)d_in[10];
  const float* W2      = (const float*)d_in[11];
  const float* b2      = (const float*)d_in[12];
  float* out = (float*)d_out;

  __bf16* ws = (__bf16*)d_ws;
  __bf16* whh_enc_frag = ws;                          // 8*64*512  = 262144 elems
  __bf16* whh_dec_frag = whh_enc_frag + 262144;       // 262144
  __bf16* w1_frag      = whh_dec_frag + 262144;       // 8*16*512  = 65536
  __bf16* wih_enc_frag = w1_frag + 65536;             // 64*512    = 32768
  __bf16* wih_dec_frag = wih_enc_frag + 32768;        // 32768  (total ~1.25 MB)

  pack_b_frag<<<262144 / 256, 256, 0, stream>>>(enc_Whh, whh_enc_frag, 64, 8, 256);
  pack_b_frag<<<262144 / 256, 256, 0, stream>>>(dec_Whh, whh_dec_frag, 64, 8, 256);
  pack_b_frag<<< 65536 / 256, 256, 0, stream>>>(W1,      w1_frag,      16, 8, 256);
  pack_wih_frag<<<32768 / 256, 256, 0, stream>>>(enc_Wih, enc_bih, enc_bhh,
                                                 wih_enc_frag, 4);
  pack_wih_frag<<<32768 / 256, 256, 0, stream>>>(dec_Wih, dec_bih, dec_bhh,
                                                 wih_dec_frag, 2);

  lstm_encdec_kernel<<<32768 / 32, 512, 0, stream>>>(
      source, whh_enc_frag, wih_enc_frag, whh_dec_frag, wih_dec_frag, w1_frag,
      b1, W2, b2, out);
}
